// MatchNet_29231547417098
// MI455X (gfx1250) — matrix operand order repaired and checked by reading, not compiled, not run on hardware
//
#include <hip/hip_runtime.h>
#include <hip/hip_bf16.h>

typedef __attribute__((ext_vector_type(16))) _Float16 v16h;
typedef __attribute__((ext_vector_type(8)))  float    v8f;
typedef __attribute__((ext_vector_type(4)))  int      v4i;

// ---------------------------------------------------------------------------
// Fragment helpers (wave32, 16x16x32 f16 WMMA)
// A (16x32 f16, M x K):  lane m = lane&15, ksel = lane>=16
//   VGPR0..3 hold K = ksel*8 + 0..7   (contiguous)
//   VGPR4..7 hold K = 16 + ksel*8 + 0..7
// B (32x16 f16, K x N):  lane n = lane&15, lanes 0-15 K=0..7,16..23, lanes 16-31 K=8..15,24..31
// C/D (16x16 f32): VGPR v -> M = v + (lane>=16 ? 8 : 0), N = lane&15
// ---------------------------------------------------------------------------

__device__ inline v16h ld_b_frag(const _Float16* __restrict__ base, int stride,
                                 int row, int kt, int lane) {
    const int ksel = (lane >> 4) & 1;
    const _Float16* p = base + row * stride + kt * 32 + ksel * 8;
    v16h r;
    v4i* d = reinterpret_cast<v4i*>(&r);
    d[0] = *reinterpret_cast<const v4i*>(p);
    d[1] = *reinterpret_cast<const v4i*>(p + 16);
    return r;
}

__device__ inline v16h ld_a_frag(const _Float16* __restrict__ base, int stride,
                                 int kt, int lane) {
    const int m = lane & 15;
    const int ksel = (lane >> 4) & 1;
    const _Float16* p = base + m * stride + kt * 32 + ksel * 8;
    v16h r;
    v4i* d = reinterpret_cast<v4i*>(&r);
    d[0] = *reinterpret_cast<const v4i*>(p);        // K +0..7
    d[1] = *reinterpret_cast<const v4i*>(p + 16);   // K +16..23
    return r;
}

__device__ inline v8f wmma_f16(v16h a, v16h b, v8f c) {
    return __builtin_amdgcn_wmma_f32_16x16x32_f16(
        false, a, false, b, (short)0, c, false, false);
}

__device__ inline float sigm_(float x) { return 1.0f / (1.0f + __expf(-x)); }
__device__ inline float tanh_(float x) {
    float e = __expf(-2.0f * fabsf(x));
    float r = (1.0f - e) / (1.0f + e);
    return (x >= 0.0f) ? r : -r;
}

// ---------------------------------------------------------------------------
// Prep kernels: cast weights to f16 (fragment-ready row-major), transpose S,
// fold the two bias vectors.
// ---------------------------------------------------------------------------
__global__ void cast_f16_kernel(const float* __restrict__ src,
                                _Float16* __restrict__ dst, int n) {
    int i = blockIdx.x * 256 + threadIdx.x;
    if (i < n) dst[i] = (_Float16)src[i];
}

__global__ void transpose_cast_kernel(const float* __restrict__ S,
                                      _Float16* __restrict__ Spack) {
    // Spack[n][k] = S[k][n], n<256, k<512
    int i = blockIdx.x * 256 + threadIdx.x;   // 131072 total
    int n = i >> 9, k = i & 511;
    Spack[i] = (_Float16)S[k * 256 + n];
}

__global__ void bias_kernel(const float* __restrict__ bih,
                            const float* __restrict__ bhh,
                            float* __restrict__ bsum) {
    int i = blockIdx.x * 256 + threadIdx.x;
    if (i < 2048) bsum[i] = bih[i] + bhh[i];
}

// ---------------------------------------------------------------------------
// Main kernel: one block = 16 batch rows, 8 waves, all 4 recurrent steps.
// ---------------------------------------------------------------------------
__global__ __launch_bounds__(256, 2)
void matchnet_kernel(const float* __restrict__ qmean,
                     const _Float16* __restrict__ Wih16,   // [2048][256]
                     const _Float16* __restrict__ Whh16,   // [2048][512]
                     const _Float16* __restrict__ S16,     // [512][256]
                     const _Float16* __restrict__ Spack,   // [256][512] = S^T
                     const float* __restrict__ bsum,       // [2048]
                     float* __restrict__ out) {
    const int tid  = threadIdx.x;
    const int lane = tid & 31;
    const int wave = tid >> 5;
    const int m0   = blockIdx.x * 16;

    __shared__ _Float16 xA [16 * 256];   //  8 KB  query rows, f16 (static)
    __shared__ _Float16 hrA[16 * 512];   // 16 KB  h_r = [h | r], f16
    __shared__ float    Lbuf[16 * 512];  // 32 KB  gates-hnew / logits / exp
    __shared__ float    red[16 * 16];
    __shared__ float    rowmax[16];
    __shared__ float    rowinv[16];

    // ---- init: cast query tile, zero h_r -------------------------------
    #pragma unroll
    for (int i = 0; i < 16; ++i) {
        int linear = i * 256 + tid;                 // 4096 elems
        int m = linear >> 8, k = linear & 255;
        xA[m * 256 + k] = (_Float16)qmean[(m0 + m) * 256 + k];
    }
    #pragma unroll
    for (int i = 0; i < 32; ++i)
        hrA[i * 256 + tid] = (_Float16)0.0f;
    __syncthreads();

    // cell state lives in registers: 4 j-tiles per wave, whole kernel
    v8f cfrag[4];
    #pragma unroll
    for (int q = 0; q < 4; ++q)
        #pragma unroll
        for (int v = 0; v < 8; ++v) cfrag[q][v] = 0.0f;

    for (int step = 0; step < 4; ++step) {
        // Memory-level laundering: blocks LICM from hoisting the (loop-
        // invariant) weight loads out of the 4-step loop -- which previously
        // spilled hundreds of VGPRs to scratch -- while keeping pointer
        // provenance intact so the backend selects global_load (not flat).
        asm volatile("" ::: "memory");

        // ================= Phase 1: LSTM gates + cell update ============
        #pragma unroll
        for (int q = 0; q < 4; ++q) {
            const int jt = wave + q * 8;            // hidden col tile 0..31
            const int n  = jt * 16 + (lane & 15);   // this lane's gate col
            v8f ai, af, ag, ao;
            #pragma unroll
            for (int v = 0; v < 8; ++v) { ai[v]=0.f; af[v]=0.f; ag[v]=0.f; ao[v]=0.f; }

            for (int kt = 0; kt < 8; ++kt) {        // x @ W_ih^T  (K = 256)
                v16h a = ld_a_frag(xA, 256, kt, lane);
                ai = wmma_f16(a, ld_b_frag(Wih16, 256,        n, kt, lane), ai);
                af = wmma_f16(a, ld_b_frag(Wih16, 256,  512 + n, kt, lane), af);
                ag = wmma_f16(a, ld_b_frag(Wih16, 256, 1024 + n, kt, lane), ag);
                ao = wmma_f16(a, ld_b_frag(Wih16, 256, 1536 + n, kt, lane), ao);
            }
            for (int kt = 0; kt < 16; ++kt) {       // h_r @ W_hh^T (K = 512)
                v16h a = ld_a_frag(hrA, 512, kt, lane);
                ai = wmma_f16(a, ld_b_frag(Whh16, 512,        n, kt, lane), ai);
                af = wmma_f16(a, ld_b_frag(Whh16, 512,  512 + n, kt, lane), af);
                ag = wmma_f16(a, ld_b_frag(Whh16, 512, 1024 + n, kt, lane), ag);
                ao = wmma_f16(a, ld_b_frag(Whh16, 512, 1536 + n, kt, lane), ao);
            }
            const float bi = bsum[n],        bf = bsum[512 + n];
            const float bg = bsum[1024 + n], bo = bsum[1536 + n];
            const int mbase = (lane >> 4) << 3;
            #pragma unroll
            for (int v = 0; v < 8; ++v) {
                float iv = sigm_(ai[v] + bi);
                float fv = sigm_(af[v] + bf);
                float gv = tanh_(ag[v] + bg);
                float ov = sigm_(ao[v] + bo);
                float cn = fv * cfrag[q][v] + iv * gv;
                cfrag[q][v] = cn;
                if (jt < 16)                         // only h_new[:, :256] used
                    Lbuf[(v + mbase) * 512 + n] = ov * tanh_(cn);
            }
        }
        __syncthreads();

        // ================= Phase 2: h = query + h_new -> hrA[:, :256] ===
        #pragma unroll
        for (int i = 0; i < 16; ++i) {
            int linear = i * 256 + tid;
            int m = linear >> 8, k = linear & 255;
            float h = qmean[(m0 + m) * 256 + k] + Lbuf[m * 512 + k];
            hrA[m * 512 + k] = (_Float16)h;
        }
        __syncthreads();

        // ================= Phase 3: logits = h @ S^T -> Lbuf ============
        {
            v8f acc[4];
            #pragma unroll
            for (int t = 0; t < 4; ++t)
                #pragma unroll
                for (int v = 0; v < 8; ++v) acc[t][v] = 0.0f;
            for (int kt = 0; kt < 8; ++kt) {         // K = 256 (h part of hrA)
                v16h a = ld_a_frag(hrA, 512, kt, lane);
                #pragma unroll
                for (int t = 0; t < 4; ++t) {
                    int nt = wave * 4 + t;           // 32 N-tiles / 8 waves
                    acc[t] = wmma_f16(a,
                        ld_b_frag(S16, 256, nt * 16 + (lane & 15), kt, lane),
                        acc[t]);
                }
            }
            const int mbase = (lane >> 4) << 3;
            #pragma unroll
            for (int t = 0; t < 4; ++t) {
                int col = (wave * 4 + t) * 16 + (lane & 15);
                #pragma unroll
                for (int v = 0; v < 8; ++v)
                    Lbuf[(v + mbase) * 512 + col] = acc[t][v];
            }
        }
        __syncthreads();

        if (step == 3) break;   // final scores are exactly these logits

        // ================= Phase 4: softmax numerators in-place =========
        {
            const int row = tid >> 4, seg = tid & 15;
            float* Lr = &Lbuf[row * 512 + seg * 32];
            float mx = -3.4e38f;
            #pragma unroll 8
            for (int c = 0; c < 32; ++c) mx = fmaxf(mx, Lr[c]);
            red[row * 16 + seg] = mx;
            __syncthreads();
            if (tid < 16) {
                float m = red[tid * 16];
                for (int s = 1; s < 16; ++s) m = fmaxf(m, red[tid * 16 + s]);
                rowmax[tid] = m;
            }
            __syncthreads();
            float rm = rowmax[row], sum = 0.0f;
            #pragma unroll 8
            for (int c = 0; c < 32; ++c) {
                float e = __expf(Lr[c] - rm);
                Lr[c] = e;
                sum += e;
            }
            red[row * 16 + seg] = sum;
            __syncthreads();
            if (tid < 16) {
                float s = 0.0f;
                for (int t2 = 0; t2 < 16; ++t2) s += red[tid * 16 + t2];
                rowinv[tid] = 1.0f / s;
            }
            __syncthreads();
        }

        // ================= Phase 5: r = softmax @ S -> hrA[:, 256:] =====
        {
            v8f acc[2];
            #pragma unroll
            for (int t = 0; t < 2; ++t)
                #pragma unroll
                for (int v = 0; v < 8; ++v) acc[t][v] = 0.0f;
            const int m = lane & 15, ksel = (lane >> 4) & 1;
            for (int kt = 0; kt < 16; ++kt) {        // K = 512 exp weights
                const float* p = &Lbuf[m * 512 + kt * 32 + ksel * 8];
                v16h a;
                #pragma unroll
                for (int t = 0; t < 8; ++t) a[t]     = (_Float16)p[t];
                #pragma unroll
                for (int t = 0; t < 8; ++t) a[8 + t] = (_Float16)p[16 + t];
                #pragma unroll
                for (int t = 0; t < 2; ++t) {
                    int nt = wave * 2 + t;           // 16 N-tiles / 8 waves
                    acc[t] = wmma_f16(a,
                        ld_b_frag(Spack, 512, nt * 16 + (lane & 15), kt, lane),
                        acc[t]);
                }
            }
            const int mbase = (lane >> 4) << 3;
            #pragma unroll
            for (int t = 0; t < 2; ++t) {
                int col = 256 + (wave * 2 + t) * 16 + (lane & 15);
                #pragma unroll
                for (int v = 0; v < 8; ++v) {
                    int M = v + mbase;
                    hrA[M * 512 + col] = (_Float16)(acc[t][v] * rowinv[M]);
                }
            }
        }
        __syncthreads();
    }

    // ---- write final scores (step-3 logits) -----------------------------
    #pragma unroll
    for (int i = 0; i < 32; ++i) {
        int linear = i * 256 + tid;                  // 8192 f32 per block
        out[m0 * 512 + linear] = Lbuf[linear];
    }
}

// ---------------------------------------------------------------------------
extern "C" void kernel_launch(void* const* d_in, const int* in_sizes, int n_in,
                              void* d_out, int out_size, void* d_ws, size_t ws_size,
                              hipStream_t stream) {
    const float* support_mean = (const float*)d_in[0];
    const float* query_mean   = (const float*)d_in[2];
    const float* W_ih         = (const float*)d_in[4];   // [2048, 256]
    const float* W_hh         = (const float*)d_in[5];   // [2048, 512]
    const float* b_ih         = (const float*)d_in[6];   // [2048]
    const float* b_hh         = (const float*)d_in[7];   // [2048]

    char* ws = (char*)d_ws;
    _Float16* Wih16 = (_Float16*)(ws);                       // 1 MB
    _Float16* Whh16 = (_Float16*)(ws + (1u << 20));          // 2 MB
    _Float16* S16   = (_Float16*)(ws + 3u * (1u << 20));     // 256 KB
    _Float16* Spack = (_Float16*)(ws + 3u * (1u << 20) + (256u << 10));
    float*    bsum  = (float*)   (ws + 3u * (1u << 20) + (512u << 10));

    cast_f16_kernel<<<2048, 256, 0, stream>>>(W_ih, Wih16, 2048 * 256);
    cast_f16_kernel<<<4096, 256, 0, stream>>>(W_hh, Whh16, 2048 * 512);
    cast_f16_kernel<<< 512, 256, 0, stream>>>(support_mean, S16, 512 * 256);
    transpose_cast_kernel<<<512, 256, 0, stream>>>(support_mean, Spack);
    bias_kernel<<<8, 256, 0, stream>>>(b_ih, b_hh, bsum);

    const int B = in_sizes[2] / 256;                 // 32768
    matchnet_kernel<<<B / 16, 256, 0, stream>>>(
        query_mean, Wih16, Whh16, S16, Spack, bsum, (float*)d_out);
}
